// MOELayerCustom_57337813402154
// MI455X (gfx1250) — matrix operand order repaired and checked by reading, not compiled, not run on hardware
//
#include <hip/hip_runtime.h>
#include <cstdint>
#include <cstddef>

#define NEXP   8
#define DMODEL 1024
#define HDIM   2048
#define NTOK   8192   // B*S = 4*2048
#define TTILE  32     // tokens per expert tile (2 WMMA n-tiles)

typedef __attribute__((ext_vector_type(16))) _Float16     v16h;
typedef __attribute__((ext_vector_type(8)))  float        v8f;
typedef __attribute__((ext_vector_type(4)))  unsigned int u32x4;
typedef __attribute__((ext_vector_type(2)))  unsigned int u32x2;
typedef __attribute__((ext_vector_type(4)))  float        f32x4;

union FragH { v16h h; u32x4 q[2]; };

static __device__ __forceinline__ v8f wmma_f16(v16h a, v16h b, v8f c) {
  // D = A(16x32 f16) * B(32x16 f16) + C(16x16 f32)
  return __builtin_amdgcn_wmma_f32_16x16x32_f16(false, a, false, b, (short)0, c,
                                                false, false);
}

// ---------------------------------------------------------------------------
// 0) zero the per-expert counters
// ---------------------------------------------------------------------------
__global__ void moe_zero_counts(int* counts) {
  if (threadIdx.x < NEXP) counts[threadIdx.x] = 0;
}

// ---------------------------------------------------------------------------
// 1) transpose + fp32->fp16 convert:  src [E][R][C] f32  ->  dst [E][C][R] f16
//    grid: (C/32, R/32, E), block (32, 8)
// ---------------------------------------------------------------------------
__global__ __launch_bounds__(256) void moe_transpose_cvt(
    const float* __restrict__ src, _Float16* __restrict__ dst, int R, int C) {
  __shared__ float tile[32][33];
  const int e  = blockIdx.z;
  const float* s = src + (size_t)e * R * C;
  _Float16*    d = dst + (size_t)e * R * C;
  const int c0 = blockIdx.x * 32, r0 = blockIdx.y * 32;
  const int tx = threadIdx.x, ty = threadIdx.y;
  #pragma unroll
  for (int i = 0; i < 32; i += 8)
    tile[ty + i][tx] = s[(size_t)(r0 + ty + i) * C + (c0 + tx)];
  __syncthreads();
  #pragma unroll
  for (int i = 0; i < 32; i += 8)
    d[(size_t)(c0 + ty + i) * R + (r0 + tx)] = (_Float16)tile[tx][ty + i];
}

// ---------------------------------------------------------------------------
// 2) router: top-2 softmax gating + per-expert gather-list build
//    one thread per token; grid NTOK/128, block 128
// ---------------------------------------------------------------------------
__global__ __launch_bounds__(128) void moe_router(
    const float* __restrict__ x, const float* __restrict__ Wg,
    int* __restrict__ counts, int* __restrict__ slotList,
    float* __restrict__ slotW) {
  const int t = blockIdx.x * 128 + threadIdx.x;
  if (t >= NTOK) return;
  const float* xr = x + (size_t)t * DMODEL;

  float acc[NEXP];
  #pragma unroll
  for (int e = 0; e < NEXP; ++e) acc[e] = 0.f;

  for (int d0 = 0; d0 < DMODEL; d0 += 4) {
    f32x4 v = *(const f32x4*)(xr + d0);
    #pragma unroll
    for (int e = 0; e < NEXP; ++e)
      acc[e] += v.x * Wg[(d0 + 0) * NEXP + e] + v.y * Wg[(d0 + 1) * NEXP + e] +
                v.z * Wg[(d0 + 2) * NEXP + e] + v.w * Wg[(d0 + 3) * NEXP + e];
  }
  // top-1
  int e1 = 0; float l1 = acc[0];
  #pragma unroll
  for (int e = 1; e < NEXP; ++e) if (acc[e] > l1) { l1 = acc[e]; e1 = e; }
  // top-2
  int e2 = (e1 == 0) ? 1 : 0; float l2 = acc[e2];
  #pragma unroll
  for (int e = 0; e < NEXP; ++e)
    if (e != e1 && acc[e] > l2) { l2 = acc[e]; e2 = e; }

  // normalized-over-top-k softmax weights: ratio of exps
  const float a  = __expf(l2 - l1);          // <= 1
  const float w0 = 1.f / (1.f + a);
  const float w1 = a * w0;
  slotW[2 * t + 0] = w0;
  slotW[2 * t + 1] = w1;

  int p = atomicAdd(&counts[e1], 1);
  slotList[e1 * NTOK + p] = 2 * t + 0;
  p = atomicAdd(&counts[e2], 1);
  slotList[e2 * NTOK + p] = 2 * t + 1;
}

// ---------------------------------------------------------------------------
// 3) expert MLP on gathered 32-token tiles.
//    grid (NTOK/TTILE, NEXP), block 256 (8 waves of 32)
//    layer1: hT[h,tok]  = relu(W1T[h,:] . xT[:,tok] + b1)   (A = weights, global)
//    layer2: yT[do,tok] += W2T[do,:]    . hT[:,tok]         (A = weights, global)
//    Each A fragment feeds 2 token n-tiles; each B fragment feeds 2/8 m-tiles.
// ---------------------------------------------------------------------------
__global__ __launch_bounds__(256) void moe_expert(
    const float* __restrict__ x, const _Float16* __restrict__ W1T,
    const _Float16* __restrict__ W2T, const float* __restrict__ b1,
    const float* __restrict__ b2, const int* __restrict__ counts,
    const int* __restrict__ slotList, float* __restrict__ yslot) {
  const int e        = blockIdx.y;
  const int tileBase = blockIdx.x * TTILE;
  const int cnt      = counts[e];
  if (tileBase >= cnt) return;
  const int rows = min(TTILE, cnt - tileBase);

  __shared__ __align__(16) _Float16 xlds[TTILE][DMODEL + 8]; // xT feed (row-major x)
  __shared__ __align__(16) _Float16 hlds[TTILE][256 + 8];    // h chunk, tok-major
  __shared__ int slots[TTILE];

  const int tid  = threadIdx.x;
  const int wave = tid >> 5;
  const int lane = tid & 31;
  const int hi   = lane >> 4;          // lane half
  const int ncol = lane & 15;          // B/C/D column within an n-tile
  const int kA   = hi * 8;             // AMD table-order K base (unused)  {0..7,16..23}/{8..15,24..31}
  const int kB   = hi * 16;            // K base for A and B (contiguous)  {0..15}/{16..31}

  if (tid < TTILE)
    slots[tid] = (tid < rows) ? slotList[e * NTOK + tileBase + tid] : 0;
  __syncthreads();

  // ---- gather + convert the 32 token rows into LDS (f16) ----
  {
    const int row = tid >> 3;            // 32 rows
    const int seg = tid & 7;             // 8 x 128-col segments
    const int tok = slots[row] >> 1;
    const float* xr = x + (size_t)tok * DMODEL + seg * 128;
    #pragma unroll
    for (int c = 0; c < 128; c += 4) {
      f32x4 v = *(const f32x4*)(xr + c);
      union { _Float16 h[4]; u32x2 u; } p;
      p.h[0] = (_Float16)v.x; p.h[1] = (_Float16)v.y;
      p.h[2] = (_Float16)v.z; p.h[3] = (_Float16)v.w;
      *(u32x2*)&xlds[row][seg * 128 + c] = p.u;
    }
  }
  __syncthreads();

  v8f yacc[8][2];
  #pragma unroll
  for (int m = 0; m < 8; ++m) {
    yacc[m][0] = (v8f)0.f;
    yacc[m][1] = (v8f)0.f;
  }

  for (int it = 0; it < HDIM / 256; ++it) {
    const int hbase = it * 256;

    // ---- layer 1: this wave produces h columns [hbase+wave*32, +32) ----
    {
      v8f hacc[2][2];
      #pragma unroll
      for (int j = 0; j < 2; ++j) {
        hacc[j][0] = (v8f)0.f;
        hacc[j][1] = (v8f)0.f;
      }
      const int hrow0 = hbase + wave * 32 + ncol;           // A row of m-tile 0
      const _Float16* A0 = W1T + ((size_t)e * HDIM + hrow0) * DMODEL;
      const _Float16* A1 = A0 + (size_t)16 * DMODEL;        // m-tile 1
      for (int kc = 0; kc < DMODEL; kc += 32) {
        __builtin_prefetch(A0 + kc + 512, 0, 1);
        __builtin_prefetch(A1 + kc + 512, 0, 1);
        FragH a0, a1, b0, b1f;
        const _Float16* Bp0 = &xlds[ncol][kc + kB];
        const _Float16* Bp1 = &xlds[16 + ncol][kc + kB];
        b0.q[0]  = *(const u32x4*)(Bp0);
        b0.q[1]  = *(const u32x4*)(Bp0 + 8);
        b1f.q[0] = *(const u32x4*)(Bp1);
        b1f.q[1] = *(const u32x4*)(Bp1 + 8);
        a0.q[0]  = *(const u32x4*)(A0 + kc + kB);
        a0.q[1]  = *(const u32x4*)(A0 + kc + kB + 8);
        a1.q[0]  = *(const u32x4*)(A1 + kc + kB);
        a1.q[1]  = *(const u32x4*)(A1 + kc + kB + 8);
        hacc[0][0] = wmma_f16(a0.h, b0.h,  hacc[0][0]);  // shared A & B reuse
        hacc[0][1] = wmma_f16(a0.h, b1f.h, hacc[0][1]);
        hacc[1][0] = wmma_f16(a1.h, b0.h,  hacc[1][0]);
        hacc[1][1] = wmma_f16(a1.h, b1f.h, hacc[1][1]);
      }
      // bias + relu + f16 store (hlds[tok][h_local] == layer-2 B layout)
      const int hl0 = wave * 32 + hi * 8;
      #pragma unroll
      for (int j = 0; j < 2; ++j) {
        #pragma unroll
        for (int r = 0; r < 8; ++r) {
          const float bb = b1[e * HDIM + hbase + hl0 + j * 16 + r];
          float v0 = hacc[j][0][r] + bb;
          float v1 = hacc[j][1][r] + bb;
          v0 = v0 > 0.f ? v0 : 0.f;
          v1 = v1 > 0.f ? v1 : 0.f;
          hlds[ncol][hl0 + j * 16 + r]      = (_Float16)v0;
          hlds[16 + ncol][hl0 + j * 16 + r] = (_Float16)v1;
        }
      }
    }
    __syncthreads();

    // ---- layer 2: this wave owns dout columns [wave*128, +128) ----
    {
      const _Float16* W2base =
          W2T + ((size_t)e * DMODEL + wave * 128 + ncol) * HDIM + hbase;
      #pragma unroll
      for (int kc = 0; kc < 256; kc += 32) {
        FragH b0, b1f;
        const _Float16* Bp0 = &hlds[ncol][kc + kB];
        const _Float16* Bp1 = &hlds[16 + ncol][kc + kB];
        b0.q[0]  = *(const u32x4*)(Bp0);
        b0.q[1]  = *(const u32x4*)(Bp0 + 8);
        b1f.q[0] = *(const u32x4*)(Bp1);
        b1f.q[1] = *(const u32x4*)(Bp1 + 8);
        #pragma unroll
        for (int mt = 0; mt < 8; ++mt) {   // one A fragment -> 2 WMMAs
          FragH a;
          const _Float16* Arow = W2base + (size_t)mt * 16 * HDIM + kc;
          a.q[0] = *(const u32x4*)(Arow + kB);
          a.q[1] = *(const u32x4*)(Arow + kB + 8);
          yacc[mt][0] = wmma_f16(a.h, b0.h,  yacc[mt][0]);
          yacc[mt][1] = wmma_f16(a.h, b1f.h, yacc[mt][1]);
        }
      }
    }
    __syncthreads();   // protect hlds before next iteration overwrites it
  }

  // ---- epilogue: + b2, scatter yT tile to per-slot buffer ----
  #pragma unroll
  for (int p = 0; p < 2; ++p) {
    const int trow  = p * 16 + ncol;
    const bool valid = trow < rows;
    const int  slot  = slots[trow];
    #pragma unroll
    for (int mt = 0; mt < 8; ++mt) {
      const int dbase = wave * 128 + mt * 16 + hi * 8;
      #pragma unroll
      for (int r = 0; r < 8; ++r) {
        const int dout = dbase + r;
        const float v = yacc[mt][p][r] + b2[e * DMODEL + dout];
        if (valid) yslot[(size_t)slot * DMODEL + dout] = v;
      }
    }
  }
}

// ---------------------------------------------------------------------------
// 4) combine: out[t] = w0*y[2t] + w1*y[2t+1]   (deterministic, atomic-free)
//    grid NTOK*DMODEL/4/256, block 256 (4 floats per thread)
// ---------------------------------------------------------------------------
__global__ __launch_bounds__(256) void moe_combine(
    const float* __restrict__ yslot, const float* __restrict__ slotW,
    float* __restrict__ out) {
  const int gid = blockIdx.x * 256 + threadIdx.x;
  const int t   = gid >> 8;                 // DMODEL/4 = 256 threads per token
  const int dq  = (gid & 255) * 4;
  const float w0 = slotW[2 * t + 0];
  const float w1 = slotW[2 * t + 1];
  const f32x4 y0 = *(const f32x4*)(yslot + (size_t)(2 * t + 0) * DMODEL + dq);
  const f32x4 y1 = *(const f32x4*)(yslot + (size_t)(2 * t + 1) * DMODEL + dq);
  f32x4 o = w0 * y0 + w1 * y1;
  *(f32x4*)(out + (size_t)t * DMODEL + dq) = o;
}

// ---------------------------------------------------------------------------
extern "C" void kernel_launch(void* const* d_in, const int* in_sizes, int n_in,
                              void* d_out, int out_size, void* d_ws, size_t ws_size,
                              hipStream_t stream) {
  (void)in_sizes; (void)n_in; (void)out_size; (void)ws_size;
  const float* x  = (const float*)d_in[0];
  const float* Wg = (const float*)d_in[1];
  const float* W1 = (const float*)d_in[2];
  const float* b1 = (const float*)d_in[3];
  const float* W2 = (const float*)d_in[4];
  const float* b2 = (const float*)d_in[5];
  float* out = (float*)d_out;

  // workspace layout (all 256-byte aligned)
  char* ws = (char*)d_ws;
  size_t o = 0;
  int* counts = (int*)(ws + o);                 o += 256;
  int* slotList = (int*)(ws + o);               o += (size_t)NEXP * NTOK * 4;
  float* slotW = (float*)(ws + o);              o += (size_t)2 * NTOK * 4;
  _Float16* W1T = (_Float16*)(ws + o);          o += (size_t)NEXP * DMODEL * HDIM * 2;
  _Float16* W2T = (_Float16*)(ws + o);          o += (size_t)NEXP * DMODEL * HDIM * 2;
  float* yslot = (float*)(ws + o);              o += (size_t)2 * NTOK * DMODEL * 4;

  moe_zero_counts<<<1, 32, 0, stream>>>(counts);
  // W1 [E][D][H] -> W1T [E][H][D]
  moe_transpose_cvt<<<dim3(HDIM / 32, DMODEL / 32, NEXP), dim3(32, 8), 0, stream>>>(
      W1, W1T, DMODEL, HDIM);
  // W2 [E][H][D] -> W2T [E][D][H]
  moe_transpose_cvt<<<dim3(DMODEL / 32, HDIM / 32, NEXP), dim3(32, 8), 0, stream>>>(
      W2, W2T, HDIM, DMODEL);
  moe_router<<<NTOK / 128, 128, 0, stream>>>(x, Wg, counts, slotList, slotW);
  moe_expert<<<dim3(NTOK / TTILE, NEXP), 256, 0, stream>>>(
      x, W1T, W2T, b1, b2, counts, slotList, yslot);
  moe_combine<<<(NTOK * DMODEL / 4) / 256, 256, 0, stream>>>(yslot, slotW, out);
}
